// LocalWindowAttention_80711025427186
// MI455X (gfx1250) — matrix operand order repaired and checked by reading, not compiled, not run on hardware
//
#include <hip/hip_runtime.h>
#include <hip/hip_bf16.h>
#include <stdint.h>

// ---------------------------------------------------------------------------
// LocalWindowAttention (EfficientViT cascaded group attention) for MI455X.
// One workgroup per 8x8 window; all GEMMs on v_wmma_f32_16x16x32_bf16.
// ---------------------------------------------------------------------------

typedef __bf16 bf16;
typedef __attribute__((ext_vector_type(16))) __bf16 v16bf;
typedef __attribute__((ext_vector_type(8)))  float  v8f;

#define B_    64
#define C_    256
#define RES_  64
#define NH_   4
#define N_    64
#define EPS_  1e-5f

// LDS strides (elements). Odd dword strides to avoid bank conflicts.
#define WA_LD   66   // bf16 A (weights) row stride
#define QKV_LD  65   // f32 q/k/v row stride
#define FEAT_LD 66   // bf16 feat (B^T) row stride
#define CC_LD   264  // bf16 concat (B^T) row stride
#define PA_LD   66   // bf16 p_w chunk (A) row stride

struct SharedP1 {
  alignas(16) bf16  wA[192 * WA_LD];     // stacked Wq/Wk/Wv, [m][k] bf16
  alignas(16) float qkv[192 * QKV_LD];   // BN'd Q(0-63) K(64-127) V(128-191), [c][n]
  alignas(16) float qc[64 * QKV_LD];     // dwconv output (BN'd Q), [c][n]
  alignas(16) float dww[64 * 25];        // depthwise weights
  float scb[192], bsb[192];              // q/k/v BN scale/bias
  float scd[64],  bsd[64];               // dw BN scale/bias
  float red[512];                        // softmax partial max/sum
};
struct SharedP2 {
  alignas(16) bf16  pA[256 * PA_LD];     // p_w K-chunk, [m][k] bf16
  float scp[256], bsp[256];              // projection BN scale/bias
};
struct Shared {
  union { SharedP1 p1; SharedP2 p2; };
  alignas(16) bf16  featB[64 * FEAT_LD];   // feat transposed [n][d] bf16 (B operand)
  alignas(16) float prevOut[64 * QKV_LD];  // cascade buffer [n][c] f32
  alignas(16) bf16  concatB[64 * CC_LD];   // head outputs [n][head*64+c] bf16 (B operand)
};

union Frag { v16bf v; uint32_t u[8]; };

// A fragment: rows m0..m0+15, K k0..k0+31 from row-major [M][ld] bf16 tile.
// 16-bit A 16x32 layout: lane h*16+r holds row r; V0..V3: K=2v+p (+8h), V4..V7: K=16+2(v-4)+p (+8h)
static __device__ __forceinline__ v16bf load_a(const bf16* base, int ld, int m0, int k0,
                                               int h, int r) {
  Frag f;
  const bf16* p = base + (m0 + r) * ld + k0 + 8 * h;
  #pragma unroll
  for (int j = 0; j < 8; ++j) {
    int koff = 2 * j + (j >= 4 ? 8 : 0);
    f.u[j] = *(const uint32_t*)(p + koff);
  }
  return f.v;
}

// B fragment from TRANSPOSED storage [n][ld] (k contiguous per lane).
// 16-bit B 32x16 layout: lane h*16+r holds column r; V0..V3: K=2v+p (+8h), V4..V7: K=16+2(v-4)+p (+8h)
static __device__ __forceinline__ v16bf load_b(const bf16* baseT, int ld, int k0, int n0,
                                               int h, int r) {
  Frag f;
  const bf16* p = baseT + (n0 + r) * ld + k0 + 8 * h;
  #pragma unroll
  for (int j = 0; j < 8; ++j)
    f.u[j] = *(const uint32_t*)(p + 2 * j + (j >= 4 ? 8 : 0));
  return f.v;
}

static __device__ __forceinline__ v8f zero8() {
  v8f z;
  #pragma unroll
  for (int i = 0; i < 8; ++i) z[i] = 0.f;
  return z;
}

__global__ __launch_bounds__(256)
void win_attn_kernel(const float* __restrict__ x,
                     const float* __restrict__ q_w,  const float* __restrict__ q_bn,
                     const float* __restrict__ k_w,  const float* __restrict__ k_bn,
                     const float* __restrict__ v_w,  const float* __restrict__ v_bn,
                     const float* __restrict__ dw_w, const float* __restrict__ dw_bn,
                     const float* __restrict__ p_w,  const float* __restrict__ p_bn,
                     float* __restrict__ y) {
  __shared__ Shared sm;

  const int tid  = threadIdx.x;
  const int lane = tid & 31;
  const int wave = tid >> 5;
  const int h16  = lane >> 4;
  const int r16  = lane & 15;

  const int w    = blockIdx.x;          // window id, 0..4095
  const int b    = w >> 6;
  const int wy   = (w >> 3) & 7;
  const int wx   = w & 7;
  const int row0 = wy * 8, col0 = wx * 8;
  const float* xb = x + (size_t)b * C_ * RES_ * RES_;

  // =========================== cascaded heads ==============================
  for (int head = 0; head < NH_; ++head) {
    // ---- BN affine params for q/k/v outputs (rows 0..191) ----
    if (tid < 192) {
      int sect = tid >> 6, c = tid & 63;
      const float* bn = (sect == 0 ? q_bn : sect == 1 ? k_bn : v_bn) + head * 4 * 64;
      float g = bn[c], be = bn[64 + c], mn = bn[128 + c], vr = bn[192 + c];
      float sc = g * rsqrtf(vr + EPS_);
      sm.p1.scb[tid] = sc;
      sm.p1.bsb[tid] = be - mn * sc;
    }
    if (tid < 64) {
      const float* bn = dw_bn + head * 4 * 64;
      float g = bn[tid], be = bn[64 + tid], mn = bn[128 + tid], vr = bn[192 + tid];
      float sc = g * rsqrtf(vr + EPS_);
      sm.p1.scd[tid] = sc;
      sm.p1.bsd[tid] = be - mn * sc;
    }
    for (int e = tid; e < 64 * 25; e += 256)
      sm.p1.dww[e] = dw_w[head * 64 * 25 + e];

    // ---- stage Wq/Wk/Wv stacked (192x64) as bf16 A ----
    for (int e = tid; e < 3 * 64 * 64; e += 256) {
      int sect = e >> 12, o = (e >> 6) & 63, d = e & 63;
      const float* wsrc = (sect == 0 ? q_w : sect == 1 ? k_w : v_w) + head * 64 * 64;
      sm.p1.wA[(sect * 64 + o) * WA_LD + d] = (bf16)wsrc[o * 64 + d];
    }

    // ---- stage feat = x chunk (+ cascade) as bf16 B^T [n][d] ----
    for (int e = tid; e < 64 * 64; e += 256) {
      int d = e >> 6, n = e & 63;
      int rr = row0 + (n >> 3), cc = col0 + (n & 7);
      float fv = xb[((size_t)(head * 64 + d) * RES_ + rr) * RES_ + cc];
      if (head > 0) fv += sm.prevOut[n * QKV_LD + d];
      sm.featB[n * FEAT_LD + d] = (bf16)fv;
    }
    __syncthreads();

    // ---- QKV GEMM: (192x64) x (64x64) on WMMA, fused BN ----
    #pragma unroll
    for (int t = 0; t < 6; ++t) {
      int tile = wave * 6 + t;
      int mt = tile >> 2, nt = tile & 3;
      v8f acc = zero8();
      #pragma unroll
      for (int ks = 0; ks < 2; ++ks) {
        v16bf a = load_a(sm.p1.wA, WA_LD, mt * 16, ks * 32, h16, r16);
        v16bf bb = load_b(sm.featB, FEAT_LD, ks * 32, nt * 16, h16, r16);
        acc = __builtin_amdgcn_wmma_f32_16x16x32_bf16(false, a, false, bb,
                                                      (short)0, acc, false, false);
      }
      int col = nt * 16 + r16;
      #pragma unroll
      for (int g = 0; g < 8; ++g) {
        int rown = mt * 16 + g + 8 * h16;
        sm.p1.qkv[rown * QKV_LD + col] = acc[g] * sm.p1.scb[rown] + sm.p1.bsb[rown];
      }
    }
    __syncthreads();

    // ---- depthwise 5x5 conv on Q (zero pad at window edge) + BN ----
    for (int e = tid; e < 64 * 64; e += 256) {
      int c = e >> 6, n = e & 63;
      int yy = n >> 3, xx = n & 7;
      float accv = 0.f;
      const float* wdw = &sm.p1.dww[c * 25];
      #pragma unroll
      for (int dy = 0; dy < 5; ++dy) {
        int sy = yy + dy - 2;
        if ((unsigned)sy < 8u) {
          #pragma unroll
          for (int dx = 0; dx < 5; ++dx) {
            int sx = xx + dx - 2;
            if ((unsigned)sx < 8u)
              accv += wdw[dy * 5 + dx] * sm.p1.qkv[c * QKV_LD + sy * 8 + sx];
          }
        }
      }
      sm.p1.qc[c * QKV_LD + n] = accv * sm.p1.scd[c] + sm.p1.bsd[c];
    }
    __syncthreads();

    // ---- per-token softmax over channels, scale V, cascade + concat ----
    {
      int n = tid & 63, part = tid >> 6;     // 4 threads per token, 16 ch each
      int cbase = part * 16;
      float s[16];
      float smax = -3.4e38f;
      #pragma unroll
      for (int j = 0; j < 16; ++j) {
        int c = cbase + j;
        float sv = sm.p1.qc[c * QKV_LD + n] * sm.p1.qkv[(64 + c) * QKV_LD + n];
        s[j] = sv;
        smax = fmaxf(smax, sv);
      }
      sm.p1.red[part * 64 + n] = smax;
      __syncthreads();
      float m = fmaxf(fmaxf(sm.p1.red[n], sm.p1.red[64 + n]),
                      fmaxf(sm.p1.red[128 + n], sm.p1.red[192 + n]));
      float psum = 0.f;
      #pragma unroll
      for (int j = 0; j < 16; ++j) {
        float ev = __expf(s[j] - m);
        s[j] = ev;
        psum += ev;
      }
      sm.p1.red[256 + part * 64 + n] = psum;
      __syncthreads();
      float tot = sm.p1.red[256 + n] + sm.p1.red[256 + 64 + n] +
                  sm.p1.red[256 + 128 + n] + sm.p1.red[256 + 192 + n];
      float rinv = 1.0f / tot;
      #pragma unroll
      for (int j = 0; j < 16; ++j) {
        int c = cbase + j;
        float o = sm.p1.qkv[(128 + c) * QKV_LD + n] * s[j] * rinv;
        sm.prevOut[n * QKV_LD + c] = o;                       // cascade (f32)
        sm.concatB[n * CC_LD + head * 64 + c] = (bf16)o;      // projection B^T
      }
    }
    __syncthreads();
  }

  // ===================== final projection (256x256)x(256x64) ==============
  {
    float g = p_bn[tid], be = p_bn[256 + tid], mn = p_bn[512 + tid], vr = p_bn[768 + tid];
    float sc = g * rsqrtf(vr + EPS_);
    sm.p2.scp[tid] = sc;
    sm.p2.bsp[tid] = be - mn * sc;
  }

  v8f acc[8];
  #pragma unroll
  for (int t = 0; t < 8; ++t) acc[t] = zero8();

  for (int kc = 0; kc < 4; ++kc) {
    __syncthreads();  // prior reads of pA done (and p1/scp visible on kc==0)
    for (int e = tid; e < 256 * 64; e += 256) {
      int o = e >> 6, k = e & 63;
      sm.p2.pA[o * PA_LD + k] = (bf16)p_w[o * 256 + kc * 64 + k];
    }
    __syncthreads();
    #pragma unroll
    for (int ks = 0; ks < 2; ++ks) {
      v16bf bfrag[4];
      #pragma unroll
      for (int tn = 0; tn < 4; ++tn)
        bfrag[tn] = load_b(sm.concatB, CC_LD, kc * 64 + ks * 32, tn * 16, h16, r16);
      #pragma unroll
      for (int tm = 0; tm < 2; ++tm) {
        v16bf a = load_a(sm.p2.pA, PA_LD, (wave * 2 + tm) * 16, ks * 32, h16, r16);
        #pragma unroll
        for (int tn = 0; tn < 4; ++tn)
          acc[tm * 4 + tn] = __builtin_amdgcn_wmma_f32_16x16x32_bf16(
              false, a, false, bfrag[tn], (short)0, acc[tm * 4 + tn], false, false);
      }
    }
  }
  __syncthreads();

  // ---- fused p_bn + window-reverse scatter ----
  float* yb = y + (size_t)b * C_ * RES_ * RES_;
  #pragma unroll
  for (int tm = 0; tm < 2; ++tm) {
    #pragma unroll
    for (int tn = 0; tn < 4; ++tn) {
      int n = tn * 16 + r16;
      int rr = row0 + (n >> 3), cc = col0 + (n & 7);
      #pragma unroll
      for (int g = 0; g < 8; ++g) {
        int o = (wave * 2 + tm) * 16 + g + 8 * h16;
        float val = acc[tm * 4 + tn][g] * sm.p2.scp[o] + sm.p2.bsp[o];
        yb[((size_t)o * RES_ + rr) * RES_ + cc] = val;
      }
    }
  }
}

extern "C" void kernel_launch(void* const* d_in, const int* in_sizes, int n_in,
                              void* d_out, int out_size, void* d_ws, size_t ws_size,
                              hipStream_t stream) {
  (void)in_sizes; (void)n_in; (void)out_size; (void)d_ws; (void)ws_size;
  const float* x    = (const float*)d_in[0];
  const float* q_w  = (const float*)d_in[1];
  const float* q_bn = (const float*)d_in[2];
  const float* k_w  = (const float*)d_in[3];
  const float* k_bn = (const float*)d_in[4];
  const float* v_w  = (const float*)d_in[5];
  const float* v_bn = (const float*)d_in[6];
  const float* dw_w = (const float*)d_in[7];
  const float* dw_bn= (const float*)d_in[8];
  const float* p_w  = (const float*)d_in[9];
  const float* p_bn = (const float*)d_in[10];
  float* y = (float*)d_out;

  dim3 grid(4096), block(256);
  win_attn_kernel<<<grid, block, 0, stream>>>(x, q_w, q_bn, k_w, k_bn, v_w, v_bn,
                                              dw_w, dw_bn, p_w, p_bn, y);
}
